// HierarchicalGRU_46016279610174
// MI455X (gfx1250) — matrix operand order repaired and checked by reading, not compiled, not run on hardware
//
#include <hip/hip_runtime.h>
#include <hip/hip_bf16.h>

typedef __attribute__((ext_vector_type(16))) _Float16 v16h;
typedef __attribute__((ext_vector_type(8)))  float    v8f;

#define En 256
#define Hn 256
#define Bn 16
#define Un 40
#define Wn 50
#define G3 768      // 3H
#define G4 1024     // padded gate width: r | z | i_n | h_n
#define NT 64       // G4/16 tiles along N
#define NSEQ (Bn*Un)      // 640
#define NTOK (NSEQ*Wn)    // 32000
#define SEQ_PER_BLK 16
#define WORD_BLKS (NSEQ/SEQ_PER_BLK)   // 40
#define TPW 8       // N-tiles per wave (8 waves * 8 = 64)

// ---- ISA 7.12.2 fragment index helpers (wave32) -----------------------------
// A matrix 16x32 f16: lane = m + 16*g ; element e -> kk mapping:
//   kk in [0,8): g=0,e=kk ; [8,16): g=1,e=kk-8 ; [16,24): g=0,e=kk-8 ; [24,32): g=1,e=kk-16
__device__ __forceinline__ int afrag_idx(int m, int k) {
  int kt = k >> 5, kk = k & 31;
  int g  = (kk >> 3) & 1;
  int e  = (kk & 7) | (((kk >> 4) & 1) << 3);
  return ((kt << 5) | (m | (g << 4))) * 16 + e;
}

__device__ __forceinline__ float sigf(float x) { return 1.0f / (1.0f + __expf(-x)); }

// ---- pack [Wih|Whh] (with n-gate split) into B-fragment-major f16 ----------
// dest flat f = (((kt*64)+nt)*32+lane)*16+e ; k = kt*32 + (e>>3)*16 + (lane>>4)*8 + (e&7) ; n = nt*16 + (lane&15)
__global__ void pack_weights(const float* __restrict__ Wih, const float* __restrict__ Whh,
                             int Kx, _Float16* __restrict__ dst, int total) {
  int K = Kx + Hn;
  for (int f = blockIdx.x * blockDim.x + threadIdx.x; f < total; f += gridDim.x * blockDim.x) {
    int e    = f & 15;
    int lane = (f >> 4) & 31;
    int nt   = (f >> 9) & 63;
    int kt   = f >> 15;
    int k = (kt << 5) + ((e >> 3) << 4) + ((lane >> 4) << 3) + (e & 7);
    int n = (nt << 4) + (lane & 15);
    float v = 0.f;
    if (k < K) {
      if (n < 2 * Hn) {                 // r,z: input + hidden combined
        v = (k < Kx) ? Wih[n * Kx + k] : Whh[n * Hn + (k - Kx)];
      } else if (n < G3) {              // i_n: input only
        v = (k < Kx) ? Wih[n * Kx + k] : 0.f;
      } else {                          // h_n: hidden only (rows 512..767)
        int row = n - Hn;
        v = (k < Kx) ? 0.f : Whh[row * Hn + (k - Kx)];
      }
    }
    dst[f] = (_Float16)v;
  }
}

// ---- GEMM helper: A_frag(LDS) x Bpack(global) -> gates(LDS f16) -------------
__device__ __forceinline__ void gru_gemm(const _Float16* __restrict__ Afrag,
                                         const _Float16* __restrict__ Bp,
                                         _Float16* __restrict__ gates,
                                         int KT, int wave, int lane) {
  v8f acc[TPW];
#pragma unroll
  for (int i = 0; i < TPW; ++i)
#pragma unroll
    for (int v = 0; v < 8; ++v) acc[i][v] = 0.f;
  for (int kt = 0; kt < KT; ++kt) {
    v16h a = *(const v16h*)&Afrag[((kt << 5) | lane) * 16];
#pragma unroll
    for (int i = 0; i < TPW; ++i) {
      int nt = wave * TPW + i;
      v16h b = *(const v16h*)(Bp + (((size_t)((kt << 6) | nt) << 5) | lane) * 16);
      acc[i] = __builtin_amdgcn_wmma_f32_16x16x32_f16(false, a, false, b, (short)0,
                                                      acc[i], false, false);
    }
  }
#pragma unroll
  for (int i = 0; i < TPW; ++i) {
    int nt = wave * TPW + i;
    int n  = (nt << 4) | (lane & 15);
    int mb = (lane >> 4) << 3;
#pragma unroll
    for (int v = 0; v < 8; ++v) gates[(mb + v) * G4 + n] = (_Float16)acc[i][v];
  }
}

// ---- word-level bidirectional GRU (blocks 0..39 fwd, 40..79 bwd) ------------
__global__ void __launch_bounds__(256)
word_gru(const int* __restrict__ tokens, const float* __restrict__ emb,
         const _Float16* __restrict__ BpF, const _Float16* __restrict__ BpB,
         const float* __restrict__ bihF, const float* __restrict__ bhhF,
         const float* __restrict__ bihB, const float* __restrict__ bhhB,
         float* __restrict__ w_out) {
  __shared__ __align__(32) _Float16 Afrag[16 * 32 * 16];   // K=512 (x:256 | h:256)
  __shared__ __align__(32) _Float16 gates[Bn * G4];
  const int tid = threadIdx.x;
  const int wave = tid >> 5, lane = tid & 31;
  const int dir  = blockIdx.x / WORD_BLKS;
  const int blk  = blockIdx.x % WORD_BLKS;
  const int seq0 = blk * SEQ_PER_BLK;
  const _Float16* Bp  = dir ? BpB  : BpF;
  const float*    bih = dir ? bihB : bihF;
  const float*    bhh = dir ? bhhB : bhhF;
  const int m = tid >> 4, part = tid & 15;
  const int seq = seq0 + m;

#pragma unroll
  for (int i = 0; i < 16; ++i)                       // h0 = 0
    Afrag[afrag_idx(m, En + part * 16 + i)] = (_Float16)0.f;

  for (int step = 0; step < Wn; ++step) {
    const int t = dir ? (Wn - 1 - step) : step;
    const int tok = tokens[seq * Wn + t];
#pragma unroll
    for (int i = 0; i < 16; ++i) {                   // gather x_t
      int k = part * 16 + i;
      Afrag[afrag_idx(m, k)] = (_Float16)emb[(size_t)tok * En + k];
    }
    __syncthreads();
    gru_gemm(Afrag, Bp, gates, 16, wave, lane);
    __syncthreads();
#pragma unroll
    for (int i = 0; i < 16; ++i) {                   // gate math
      int j = part * 16 + i;
      float r  = sigf((float)gates[m * G4 + j]            + bih[j]          + bhh[j]);
      float z  = sigf((float)gates[m * G4 + Hn + j]       + bih[Hn + j]     + bhh[Hn + j]);
      float hn = (float)gates[m * G4 + 3 * Hn + j]        + bhh[2 * Hn + j];
      float ng = tanhf((float)gates[m * G4 + 2 * Hn + j]  + bih[2 * Hn + j] + r * hn);
      int hi = afrag_idx(m, En + j);
      float hold = (float)Afrag[hi];
      float hnew = (1.f - z) * ng + z * hold;
      w_out[((size_t)(seq * Wn + t)) * (2 * Hn) + dir * Hn + j] = hnew;
      Afrag[hi] = (_Float16)hnew;
    }
    __syncthreads();
  }
}

// ---- gather last valid word state per utterance -----------------------------
__global__ void gather_utt(const float* __restrict__ w_out, const int* __restrict__ w_len,
                           float* __restrict__ utt_in) {
  int row = blockIdx.x;                               // 0..639
  int wl  = w_len[row];
  const float* src = w_out + ((size_t)(row * Wn + (wl - 1))) * (2 * Hn);
  float* dst = utt_in + (size_t)row * (2 * Hn);
  for (int c = threadIdx.x; c < 2 * Hn; c += blockDim.x) dst[c] = src[c];
}

// ---- adaptive utterance GRU with segmentation gate (single block) -----------
__global__ void __launch_bounds__(256)
adaptive_gru(const float* __restrict__ utt_in, const int* __restrict__ u_len,
             const _Float16* __restrict__ Bp, const float* __restrict__ bih,
             const float* __restrict__ bhh, const float* __restrict__ aWa,
             const float* __restrict__ aBa,
             float* __restrict__ u_output, float* __restrict__ seg_mask_out,
             float* __restrict__ s_len_out, float* __restrict__ seg_input) {
  __shared__ __align__(32) _Float16 Afrag[24 * 32 * 16];  // K=768 (x:512 | h:256)
  __shared__ __align__(32) _Float16 gates[Bn * G4];
  __shared__ float partial[256];
  __shared__ float xdot[Bn];
  __shared__ int   bnd[Bn];
  __shared__ int   rank[Bn];
  const int tid = threadIdx.x, wave = tid >> 5, lane = tid & 31;
  const int m = tid >> 4, part = tid & 15;

  for (int i = tid; i < Bn * Un * Hn; i += 256) seg_input[i] = 0.f;  // zero seg buffer
  if (tid < Bn) rank[tid] = 0;
#pragma unroll
  for (int i = 0; i < 16; ++i)                       // h0 = 0 (k = 512+j)
    Afrag[afrag_idx(m, 2 * Hn + part * 16 + i)] = (_Float16)0.f;

  float h1r[16];
  for (int t = 0; t < Un; ++t) {
    const float* xrow = utt_in + (size_t)(m * Un + t) * (2 * Hn);
#pragma unroll
    for (int i = 0; i < 32; ++i) {                   // x_t (512 per row)
      int k = part * 32 + i;
      Afrag[afrag_idx(m, k)] = (_Float16)xrow[k];
    }
    if (tid < Bn) {                                  // x-part of alpha dot
      const float* xr = utt_in + (size_t)(tid * Un + t) * (2 * Hn);
      float s = 0.f;
      for (int k = 0; k < 2 * Hn; ++k) s += xr[k] * aWa[k];
      xdot[tid] = s;
    }
    __syncthreads();
    gru_gemm(Afrag, Bp, gates, 24, wave, lane);
    __syncthreads();
    float pdot = 0.f;
#pragma unroll
    for (int i = 0; i < 16; ++i) {                   // h1 + u_output + alpha partial
      int j = part * 16 + i;
      float r  = sigf((float)gates[m * G4 + j]           + bih[j]          + bhh[j]);
      float z  = sigf((float)gates[m * G4 + Hn + j]      + bih[Hn + j]     + bhh[Hn + j]);
      float hn = (float)gates[m * G4 + 3 * Hn + j]       + bhh[2 * Hn + j];
      float ng = tanhf((float)gates[m * G4 + 2 * Hn + j] + bih[2 * Hn + j] + r * hn);
      float hold = (float)Afrag[afrag_idx(m, 2 * Hn + j)];
      float h1 = (1.f - z) * ng + z * hold;
      h1r[i] = h1;
      u_output[(size_t)(m * Un + t) * Hn + j] = h1;
      pdot += h1 * aWa[2 * Hn + j];
    }
    partial[tid] = pdot;
    __syncthreads();
    if (tid < Bn) {                                  // alpha + boundary
      float hs = 0.f;
#pragma unroll
      for (int i = 0; i < 16; ++i) hs += partial[tid * 16 + i];
      float alpha = sigf(xdot[tid] + hs + aBa[0]);
      int ul = u_len[tid];
      int valid = (t < ul);
      int b = ((alpha > 0.5f) || (t == ul - 1)) && valid;
      bnd[tid] = b;
      seg_mask_out[tid * Un + t] = b ? 1.f : 0.f;
    }
    __syncthreads();
    {
      int valid = (t < u_len[m]);
      int b = bnd[m];
#pragma unroll
      for (int i = 0; i < 16; ++i) {                 // reset / carry / seg scatter
        int j = part * 16 + i;
        int hi = afrag_idx(m, 2 * Hn + j);
        float hold = (float)Afrag[hi];
        float hnext = valid ? (b ? 0.f : h1r[i]) : hold;
        if (b) seg_input[(size_t)(m * Un + rank[m]) * Hn + j] = h1r[i];
        Afrag[hi] = (_Float16)hnext;
      }
    }
    __syncthreads();
    if (tid < Bn && bnd[tid]) rank[tid]++;
    __syncthreads();
  }
  if (tid < Bn) s_len_out[tid] = (float)rank[tid];
}

// ---- segment-level GRU (single block) --------------------------------------
__global__ void __launch_bounds__(256)
segment_gru(const float* __restrict__ seg_input, const _Float16* __restrict__ Bp,
            const float* __restrict__ bih, const float* __restrict__ bhh,
            float* __restrict__ s_output) {
  __shared__ __align__(32) _Float16 Afrag[16 * 32 * 16];  // K=512 (x:256 | h:256)
  __shared__ __align__(32) _Float16 gates[Bn * G4];
  const int tid = threadIdx.x, wave = tid >> 5, lane = tid & 31;
  const int m = tid >> 4, part = tid & 15;
#pragma unroll
  for (int i = 0; i < 16; ++i)
    Afrag[afrag_idx(m, Hn + part * 16 + i)] = (_Float16)0.f;

  for (int t = 0; t < Un; ++t) {
    const float* xrow = seg_input + (size_t)(m * Un + t) * Hn;
#pragma unroll
    for (int i = 0; i < 16; ++i) {
      int k = part * 16 + i;
      Afrag[afrag_idx(m, k)] = (_Float16)xrow[k];
    }
    __syncthreads();
    gru_gemm(Afrag, Bp, gates, 16, wave, lane);
    __syncthreads();
#pragma unroll
    for (int i = 0; i < 16; ++i) {
      int j = part * 16 + i;
      float r  = sigf((float)gates[m * G4 + j]           + bih[j]          + bhh[j]);
      float z  = sigf((float)gates[m * G4 + Hn + j]      + bih[Hn + j]     + bhh[Hn + j]);
      float hn = (float)gates[m * G4 + 3 * Hn + j]       + bhh[2 * Hn + j];
      float ng = tanhf((float)gates[m * G4 + 2 * Hn + j] + bih[2 * Hn + j] + r * hn);
      int hi = afrag_idx(m, Hn + j);
      float hold = (float)Afrag[hi];
      float hnew = (1.f - z) * ng + z * hold;
      s_output[(size_t)(m * Un + t) * Hn + j] = hnew;
      Afrag[hi] = (_Float16)hnew;
    }
    __syncthreads();
  }
}

// ---- prefix sums / lengths / utt_indices ------------------------------------
__global__ void meta_kernel(const int* __restrict__ u_len, const int* __restrict__ w_len,
                            int* __restrict__ start, int* __restrict__ wl2,
                            float* __restrict__ w2_len_out, float* __restrict__ utt_idx_out) {
  int b = threadIdx.x;
  if (b < Bn) {
    int cum = 0, ul = u_len[b];
    for (int u = 0; u < Un; ++u) {
      int wl = (u < ul) ? w_len[b * Un + u] : 0;
      start[b * Un + u] = cum;
      wl2[b * Un + u]   = wl;
      utt_idx_out[b * Un + u] = (u < ul) ? (float)(cum + wl) : 0.f;
      cum += wl;
    }
    w2_len_out[b] = (float)cum;
  }
}

__global__ void zero_w2(float* __restrict__ w2, size_t n) {
  size_t i = (size_t)blockIdx.x * blockDim.x + threadIdx.x;
  size_t stride = (size_t)gridDim.x * blockDim.x;
  for (; i < n; i += stride) w2[i] = 0.f;
}

__global__ void scatter_w2(const float* __restrict__ w_out, const int* __restrict__ start,
                           const int* __restrict__ wl2, float* __restrict__ w2) {
  int row = blockIdx.x;                               // 0..31999  (= b*U*W + u*W + w)
  int b  = row / (Un * Wn);
  int r2 = row % (Un * Wn);
  int u = r2 / Wn, w = r2 % Wn;
  if (w >= wl2[b * Un + u]) return;
  size_t dst = ((size_t)b * (Un * Wn) + start[b * Un + u] + w) * (2 * Hn);
  size_t src = (size_t)row * (2 * Hn);
  for (int c = threadIdx.x; c < 2 * Hn; c += blockDim.x) w2[dst + c] = w_out[src + c];
}

extern "C" void kernel_launch(void* const* d_in, const int* in_sizes, int n_in,
                              void* d_out, int out_size, void* d_ws, size_t ws_size,
                              hipStream_t stream) {
  const int*   tokens = (const int*)d_in[0];
  const int*   u_len  = (const int*)d_in[1];
  const int*   w_len  = (const int*)d_in[2];
  const float* emb    = (const float*)d_in[3];
  const float* wf_Wih = (const float*)d_in[4];
  const float* wf_Whh = (const float*)d_in[5];
  const float* wf_bih = (const float*)d_in[6];
  const float* wf_bhh = (const float*)d_in[7];
  const float* wb_Wih = (const float*)d_in[8];
  const float* wb_Whh = (const float*)d_in[9];
  const float* wb_bih = (const float*)d_in[10];
  const float* wb_bhh = (const float*)d_in[11];
  const float* a_Wih  = (const float*)d_in[12];
  const float* a_Whh  = (const float*)d_in[13];
  const float* a_bih  = (const float*)d_in[14];
  const float* a_bhh  = (const float*)d_in[15];
  const float* a_Wa   = (const float*)d_in[16];
  const float* a_ba   = (const float*)d_in[17];
  const float* s_Wih  = (const float*)d_in[18];
  const float* s_Whh  = (const float*)d_in[19];
  const float* s_bih  = (const float*)d_in[20];
  const float* s_bhh  = (const float*)d_in[21];

  char* ws = (char*)d_ws;
  _Float16* bp_wf = (_Float16*)(ws + 0);            // 512*1024*2 = 1 MB
  _Float16* bp_wb = (_Float16*)(ws + 1048576);      // 1 MB
  _Float16* bp_a  = (_Float16*)(ws + 2097152);      // 768*1024*2 = 1.5 MB
  _Float16* bp_s  = (_Float16*)(ws + 3670016);      // 1 MB
  float* w_out    = (float*)(ws + 4718592);         // 32000*512*4 = 65.5 MB
  float* utt_in   = (float*)(ws + 70254592);        // 640*512*4
  float* seg_in   = (float*)(ws + 71565312);        // 16*40*256*4
  int*   start    = (int*)(ws + 72220672);          // 640*4
  int*   wl2      = (int*)(ws + 72223232);          // 640*4

  float* out        = (float*)d_out;
  float* s_output   = out;                          // [16,40,256]
  float* u_output   = out + 163840;                 // [16,40,256]
  float* w2         = out + 327680;                 // [16,2000,512]
  float* s_len_o    = out + 16711680;               // [16]
  float* w2_len_o   = out + 16711696;               // [16]
  float* utt_idx_o  = out + 16711712;               // [16,40]
  float* seg_mask_o = out + 16712352;               // [16,40]

  pack_weights<<<512, 256, 0, stream>>>(wf_Wih, wf_Whh, 256, bp_wf, 512 * 1024);
  pack_weights<<<512, 256, 0, stream>>>(wb_Wih, wb_Whh, 256, bp_wb, 512 * 1024);
  pack_weights<<<768, 256, 0, stream>>>(a_Wih,  a_Whh,  512, bp_a,  768 * 1024);
  pack_weights<<<512, 256, 0, stream>>>(s_Wih,  s_Whh,  256, bp_s,  512 * 1024);

  word_gru<<<2 * WORD_BLKS, 256, 0, stream>>>(tokens, emb, bp_wf, bp_wb,
                                              wf_bih, wf_bhh, wb_bih, wb_bhh, w_out);
  gather_utt<<<NSEQ, 128, 0, stream>>>(w_out, w_len, utt_in);
  meta_kernel<<<1, 64, 0, stream>>>(u_len, w_len, start, wl2, w2_len_o, utt_idx_o);
  adaptive_gru<<<1, 256, 0, stream>>>(utt_in, u_len, bp_a, a_bih, a_bhh, a_Wa, a_ba,
                                      u_output, seg_mask_o, s_len_o, seg_in);
  segment_gru<<<1, 256, 0, stream>>>(seg_in, bp_s, s_bih, s_bhh, s_output);
  zero_w2<<<2048, 256, 0, stream>>>(w2, (size_t)Bn * Un * Wn * 2 * Hn);
  scatter_w2<<<NTOK, 128, 0, stream>>>(w_out, start, wl2, w2);
}
